// STORN_RNN_63050119905602
// MI455X (gfx1250) — matrix operand order repaired and checked by reading, not compiled, not run on hardware
//
#include <hip/hip_runtime.h>
#include <hip/hip_fp16.h>
#include <math.h>

typedef _Float16 h16;
typedef __attribute__((ext_vector_type(16))) _Float16 v16h;
typedef __attribute__((ext_vector_type(8)))  _Float16 v8h;
typedef __attribute__((ext_vector_type(8)))  float    v8f;

#define WMMA_F16(a, b, c) \
  __builtin_amdgcn_wmma_f32_16x16x32_f16(false, (a), false, (b), (short)0, (c), false, false)

// gfx1250 hardware transcendentals (guarded: fall back to libm if absent)
#if __has_builtin(__builtin_amdgcn_tanhf)
__device__ __forceinline__ float fast_tanh(float x) { return __builtin_amdgcn_tanhf(x); }
#else
__device__ __forceinline__ float fast_tanh(float x) { return tanhf(x); }
#endif
#if __has_builtin(__builtin_amdgcn_exp2f)
__device__ __forceinline__ float fast_exp2(float x) { return __builtin_amdgcn_exp2f(x); }
#else
__device__ __forceinline__ float fast_exp2(float x) { return exp2f(x); }
#endif

static constexpr int Bdim = 64;
static constexpr int Tdim = 512;
static constexpr int Fdim = 128;
static constexpr int Hdim = 512;

// ---------------------------------------------------------------------------
// Elementwise helpers
// ---------------------------------------------------------------------------
__global__ void k_cast_f32_f16(const float* __restrict__ in, h16* __restrict__ out, int n) {
  int i = blockIdx.x * blockDim.x + threadIdx.x;
  if (i < n) out[i] = (h16)in[i];
}

__global__ void k_add_f32(const float* __restrict__ a, const float* __restrict__ b,
                          float* __restrict__ o, int n) {
  int i = blockIdx.x * blockDim.x + threadIdx.x;
  if (i < n) o[i] = a[i] + b[i];
}

// Wcomb[n*H+k] = Wih_de[n*2H+k] + Whh_de[n*H+k]   (fold decoder h-paths)
__global__ void k_combine_wde(const float* __restrict__ wih, const float* __restrict__ whh,
                              h16* __restrict__ o, int n) {
  int i = blockIdx.x * blockDim.x + threadIdx.x;
  if (i < n) {
    int r = i / Hdim, k = i % Hdim;
    o[i] = (h16)(wih[r * 2 * Hdim + k] + whh[i]);
  }
}

// z = eps * exp(lv/2) + mu  ->  eps * exp2(lv * 0.5*log2(e)) + mu
__global__ void k_reparam(const float* __restrict__ mu, const float* __restrict__ lv,
                          const float* __restrict__ eps, h16* __restrict__ z, int n) {
  int i = blockIdx.x * blockDim.x + threadIdx.x;
  if (i < n) z[i] = (h16)(fmaf(eps[i], fast_exp2(lv[i] * 0.7213475204444817f), mu[i]));
}

// ---------------------------------------------------------------------------
// WMMA fragment loaders (contiguous k order k = 16*hi + s, same for A and B)
// ---------------------------------------------------------------------------
// A (MxK 16x32, f16): lanes 0-15 row=lane, K = k0+[0..15];
//                     lanes 16-31 same rows, K = k0+[16..31].
__device__ __forceinline__ v16h load_a_global(const h16* __restrict__ base, int lda,
                                              int m0, int k0, int lm, int hi) {
  const h16* p = base + (size_t)(m0 + lm) * lda + k0 + hi * 16;
  v8h lo = *(const v8h*)p;
  v8h hh = *(const v8h*)(p + 8);
  return __builtin_shufflevector(lo, hh, 0, 1, 2, 3, 4, 5, 6, 7, 8, 9, 10, 11, 12, 13, 14, 15);
}

// B (KxN 32x16, f16) where B = W^T, W row-major [N,K]: lane's column n = lane&15,
// lanes 0-15 hold K=k0..k0+15 (contiguous in W row n), lanes 16-31 hold k0+16..k0+31.
__device__ __forceinline__ v16h load_b_global(const h16* __restrict__ W, int ldw,
                                              int n0, int k0, int lm, int hi) {
  const h16* p = W + (size_t)(n0 + lm) * ldw + k0 + hi * 16;
  return *(const v16h*)p;
}

// ---------------------------------------------------------------------------
// Batched GEMM + bias:  C[M,N] = A[M,K] @ W[N,K]^T + bias[N]
// block = 256 threads (8 waves); block tile 64(M) x 128(N); wave tile 32x32 (2x2 WMMA tiles)
// ---------------------------------------------------------------------------
__global__ __launch_bounds__(256) void k_gemm_bias(
    const h16* __restrict__ A, int lda,
    const h16* __restrict__ W, int ldw,
    const float* __restrict__ bias,
    float* __restrict__ C, int ldc, int K) {
  const int tid = threadIdx.x;
  const int wave = tid >> 5, lane = tid & 31;
  const int hi = lane >> 4, lm = lane & 15;
  const int m0 = blockIdx.x * 64 + (wave & 1) * 32;
  const int n0 = blockIdx.y * 128 + (wave >> 1) * 32;

  v8f acc[2][2] = {};
  for (int k0 = 0; k0 < K; k0 += 32) {
    v16h a[2], b[2];
#pragma unroll
    for (int i = 0; i < 2; ++i) a[i] = load_a_global(A, lda, m0 + i * 16, k0, lm, hi);
#pragma unroll
    for (int j = 0; j < 2; ++j) b[j] = load_b_global(W, ldw, n0 + j * 16, k0, lm, hi);
#pragma unroll
    for (int i = 0; i < 2; ++i)
#pragma unroll
      for (int j = 0; j < 2; ++j) acc[i][j] = WMMA_F16(a[i], b[j], acc[i][j]);
  }
  // D layout: VGPR v -> row = v + 8*hi ; col = lane&15
#pragma unroll
  for (int i = 0; i < 2; ++i)
#pragma unroll
    for (int j = 0; j < 2; ++j) {
      int col = n0 + j * 16 + lm;
      float bv = bias[col];
#pragma unroll
      for (int v = 0; v < 8; ++v) {
        int row = m0 + i * 16 + v + hi * 8;
        C[(size_t)row * ldc + col] = acc[i][j][v] + bv;
      }
    }
}

// ---------------------------------------------------------------------------
// Persistent recurrent kernel (serves encoder AND decoder):
//   h_t = tanh( h_{t-1} @ W^T + pre[:, t, :] )      h: [64, 512] f16, LDS-resident
// Single workgroup, 1024 threads = 32 waves. 128 output tiles (4 Mt x 32 Nt),
// each wave owns 1 Mt x 4 Nt; K=512 -> 16 wmma per tile per step.
// pre[t] loads are issued BEFORE the K-loop; the K-loop is FULLY unrolled
// (64 WMMAs) so the scheduler can clause weight loads far ahead of use.
// ---------------------------------------------------------------------------
__global__ __launch_bounds__(1024) void k_recurrent(
    const float* __restrict__ pre,   // [B, T, H]  (input projection + biases)
    const h16* __restrict__ W,       // [H, H] f16 (row = output unit)
    h16* __restrict__ yout) {        // [B, T, H] f16 hidden states
  __shared__ h16 hbuf[Bdim * Hdim];  // 64 KB, persistent hidden state

  const int tid = threadIdx.x;
  for (int i = tid; i < Bdim * Hdim; i += 1024) hbuf[i] = (h16)0.0f;
  __syncthreads();

  const int wave = tid >> 5, lane = tid & 31;
  const int hi = lane >> 4, lm = lane & 15;
  const int mt = wave >> 3;        // M tile 0..3  -> rows mt*16
  const int nb = (wave & 7) * 4;   // N tile base  -> 4 tiles = 64 cols

  for (int t = 0; t < Tdim; ++t) {
    // --- issue pre[t] loads early; WMMA K-loop below hides their latency ---
    float pv[4][8];
#pragma unroll
    for (int j = 0; j < 4; ++j) {
      int col = (nb + j) * 16 + lm;
#pragma unroll
      for (int v = 0; v < 8; ++v) {
        int row = mt * 16 + v + hi * 8;  // batch index
        pv[j][v] = pre[((size_t)row * Tdim + t) * Hdim + col];
      }
    }

    v8f acc[4] = {};
#pragma unroll
    for (int k0 = 0; k0 < Hdim; k0 += 32) {
      // A fragment from LDS-resident h
      const h16* ap = &hbuf[(mt * 16 + lm) * Hdim + k0 + hi * 16];
      v8h alo = *(const v8h*)(ap);
      v8h ahi = *(const v8h*)(ap + 8);
      v16h a = __builtin_shufflevector(alo, ahi, 0, 1, 2, 3, 4, 5, 6, 7,
                                       8, 9, 10, 11, 12, 13, 14, 15);
#pragma unroll
      for (int j = 0; j < 4; ++j) {
        const h16* bp = W + (size_t)((nb + j) * 16 + lm) * Hdim + k0 + hi * 16;
        v16h b = *(const v16h*)bp;
        acc[j] = WMMA_F16(a, b, acc[j]);
      }
    }

    // epilogue: add precomputed input projection, tanh (hardware V_TANH_F32)
    float hv[4][8];
#pragma unroll
    for (int j = 0; j < 4; ++j)
#pragma unroll
      for (int v = 0; v < 8; ++v) hv[j][v] = fast_tanh(acc[j][v] + pv[j][v]);

    __syncthreads();  // all waves finished reading h_{t-1}
#pragma unroll
    for (int j = 0; j < 4; ++j) {
      int col = (nb + j) * 16 + lm;
#pragma unroll
      for (int v = 0; v < 8; ++v) {
        int row = mt * 16 + v + hi * 8;
        h16 hx = (h16)hv[j][v];
        hbuf[row * Hdim + col] = hx;
        yout[((size_t)row * Tdim + t) * Hdim + col] = hx;
      }
    }
    __syncthreads();  // h_t fully published
  }
}

// ---------------------------------------------------------------------------
// Host orchestration
// ---------------------------------------------------------------------------
extern "C" void kernel_launch(void* const* d_in, const int* in_sizes, int n_in,
                              void* d_out, int out_size, void* d_ws, size_t ws_size,
                              hipStream_t stream) {
  (void)in_sizes; (void)n_in; (void)out_size; (void)ws_size;

  const float* x      = (const float*)d_in[0];
  const float* eps    = (const float*)d_in[1];
  const float* Wih_en = (const float*)d_in[2];
  const float* Whh_en = (const float*)d_in[3];
  const float* bih_en = (const float*)d_in[4];
  const float* bhh_en = (const float*)d_in[5];
  const float* W_mu   = (const float*)d_in[6];
  const float* b_mu   = (const float*)d_in[7];
  const float* W_lv   = (const float*)d_in[8];
  const float* b_lv   = (const float*)d_in[9];
  const float* Wih_de = (const float*)d_in[10];
  const float* Whh_de = (const float*)d_in[11];
  const float* bih_de = (const float*)d_in[12];
  const float* bhh_de = (const float*)d_in[13];
  const float* W_mux  = (const float*)d_in[14];
  const float* b_mux  = (const float*)d_in[15];
  const float* W_lvx  = (const float*)d_in[16];
  const float* b_lvx  = (const float*)d_in[17];

  float* out = (float*)d_out;

  const size_t BT  = (size_t)Bdim * Tdim;          // 32768
  const size_t BTH = BT * Hdim;                    // 16.7M
  const size_t BTF = BT * Fdim;                    // 4.19M

  // workspace carving (256B aligned)
  char* ws = (char*)d_ws;
  auto carve = [&](size_t bytes) -> void* {
    void* p = (void*)ws;
    ws += (bytes + 255) & ~(size_t)255;
    return p;
  };
  h16*   x_h      = (h16*)carve(BTF * 2);
  h16*   Wih_en_h = (h16*)carve((size_t)Hdim * Fdim * 2);
  h16*   Whh_en_h = (h16*)carve((size_t)Hdim * Hdim * 2);
  h16*   W_mu_h   = (h16*)carve((size_t)Hdim * Hdim * 2);
  h16*   W_lv_h   = (h16*)carve((size_t)Hdim * Hdim * 2);
  h16*   Wih_de_h = (h16*)carve((size_t)Hdim * 2 * Hdim * 2);
  h16*   Wcomb_h  = (h16*)carve((size_t)Hdim * Hdim * 2);
  h16*   W_mux_h  = (h16*)carve((size_t)Fdim * Hdim * 2);
  h16*   W_lvx_h  = (h16*)carve((size_t)Fdim * Hdim * 2);
  float* bias_en  = (float*)carve(Hdim * 4);
  float* bias_de  = (float*)carve(Hdim * 4);
  float* pre      = (float*)carve(BTH * 4);        // reused: pre_en then zpart
  h16*   y_h      = (h16*)carve(BTH * 2);
  h16*   z_h      = (h16*)carve(BTH * 2);
  h16*   o_h      = (h16*)carve(BTH * 2);

  const int TPB = 256;
  auto blocks = [&](size_t n) { return (unsigned)((n + TPB - 1) / TPB); };

  // --- stage 0: precision casts + folded weights/biases ---
  k_cast_f32_f16<<<blocks(BTF), TPB, 0, stream>>>(x, x_h, (int)BTF);
  k_cast_f32_f16<<<blocks((size_t)Hdim * Fdim), TPB, 0, stream>>>(Wih_en, Wih_en_h, Hdim * Fdim);
  k_cast_f32_f16<<<blocks((size_t)Hdim * Hdim), TPB, 0, stream>>>(Whh_en, Whh_en_h, Hdim * Hdim);
  k_cast_f32_f16<<<blocks((size_t)Hdim * Hdim), TPB, 0, stream>>>(W_mu, W_mu_h, Hdim * Hdim);
  k_cast_f32_f16<<<blocks((size_t)Hdim * Hdim), TPB, 0, stream>>>(W_lv, W_lv_h, Hdim * Hdim);
  k_cast_f32_f16<<<blocks((size_t)Hdim * 2 * Hdim), TPB, 0, stream>>>(Wih_de, Wih_de_h, Hdim * 2 * Hdim);
  k_cast_f32_f16<<<blocks((size_t)Fdim * Hdim), TPB, 0, stream>>>(W_mux, W_mux_h, Fdim * Hdim);
  k_cast_f32_f16<<<blocks((size_t)Fdim * Hdim), TPB, 0, stream>>>(W_lvx, W_lvx_h, Fdim * Hdim);
  k_combine_wde<<<blocks((size_t)Hdim * Hdim), TPB, 0, stream>>>(Wih_de, Whh_de, Wcomb_h, Hdim * Hdim);
  k_add_f32<<<blocks(Hdim), TPB, 0, stream>>>(bih_en, bhh_en, bias_en, Hdim);
  k_add_f32<<<blocks(Hdim), TPB, 0, stream>>>(bih_de, bhh_de, bias_de, Hdim);

  // --- stage 1: pre_en = x @ Wih_en^T + (bih_en+bhh_en)   [32768 x 512], K=128 ---
  k_gemm_bias<<<dim3(BT / 64, Hdim / 128), 256, 0, stream>>>(
      x_h, Fdim, Wih_en_h, Fdim, bias_en, pre, Hdim, Fdim);

  // --- stage 2: encoder recurrence (persistent, single WGP) ---
  k_recurrent<<<1, 1024, 0, stream>>>(pre, Whh_en_h, y_h);

  // --- stage 3: posterior projections -> d_out[0], d_out[BTH] ---
  k_gemm_bias<<<dim3(BT / 64, Hdim / 128), 256, 0, stream>>>(
      y_h, Hdim, W_mu_h, Hdim, b_mu, out, Hdim, Hdim);
  k_gemm_bias<<<dim3(BT / 64, Hdim / 128), 256, 0, stream>>>(
      y_h, Hdim, W_lv_h, Hdim, b_lv, out + BTH, Hdim, Hdim);

  // --- stage 4: reparameterize ---
  k_reparam<<<blocks(BTH), TPB, 0, stream>>>(out, out + BTH, eps, z_h, (int)BTH);

  // --- stage 5: zpart = z @ Wih_de[:, H:]^T + (bih_de+bhh_de)  (reuse `pre`) ---
  k_gemm_bias<<<dim3(BT / 64, Hdim / 128), 256, 0, stream>>>(
      z_h, Hdim, Wih_de_h + Hdim, 2 * Hdim, bias_de, pre, Hdim, Hdim);

  // --- stage 6: decoder recurrence with folded Wcomb = Wih_de[:,:H] + Whh_de ---
  k_recurrent<<<1, 1024, 0, stream>>>(pre, Wcomb_h, o_h);

  // --- stage 7: output heads -> d_out[2*BTH], d_out[2*BTH+BTF]  (N=128) ---
  k_gemm_bias<<<dim3(BT / 64, Fdim / 128), 256, 0, stream>>>(
      o_h, Hdim, W_mux_h, Hdim, b_mux, out + 2 * BTH, Fdim, Hdim);
  k_gemm_bias<<<dim3(BT / 64, Fdim / 128), 256, 0, stream>>>(
      o_h, Hdim, W_lvx_h, Hdim, b_lvx, out + 2 * BTH + BTF, Fdim, Hdim);
}
